// newmodel_12154757448428
// MI455X (gfx1250) — matrix operand order repaired and checked by reading, not compiled, not run on hardware
//
#include <hip/hip_runtime.h>

// MI455X / gfx1250, wave32. All matrix work routed through
// v_wmma_f32_16x16x32_f16 (f16 in, f32 accumulate). Each wave computes a
// 64(oc) x 16(pixel) output macro-tile: one B load feeds 4 independent WMMAs.

typedef __attribute__((ext_vector_type(16))) _Float16 v16h;
typedef __attribute__((ext_vector_type(8)))  float    v8f;

static __device__ __forceinline__ float silu_f(float v) {
  return v / (1.f + __expf(-v));
}

// ---------------------------------------------------------------- utilities
__global__ void k_zero(float* p, int n) {
  int i = blockIdx.x * 256 + threadIdx.x;
  if (i < n) p[i] = 0.f;
}

// Pack fp32 weights (row-major [Cout][Cin][taps], OIHW with taps=kh*kw, or a
// plain [E][D] matrix with taps=1) into the contiguous-K 16-bit A-operand
// lane layout k_conv_wmma uses (not the ISA table order): lane = 16*g + m holds row M=m; element e maps to
// K = 16*g + e.  Result: wA[tap][mt][kt][lane][16].
// MT is pre-padded to a multiple of 4 (extra tiles zero-filled).
__global__ void k_pack_a(const float* __restrict__ w, _Float16* __restrict__ dst,
                         int Cout, int Cin, int taps, int MT, int KT) {
  int total = taps * MT * KT * 512;
  int idx = blockIdx.x * 256 + threadIdx.x;
  if (idx >= total) return;
  int e    = idx & 15;
  int lane = (idx >> 4) & 31;
  int rest = idx >> 9;
  int kt = rest % KT; rest /= KT;
  int mt = rest % MT;
  int tap = rest / MT;
  int g = lane >> 4, m = lane & 15;
  int K = g * 16 + e;
  int oc = mt * 16 + m, ci = kt * 32 + K;
  float v = (oc < Cout && ci < Cin) ? w[((size_t)oc * Cin + ci) * taps + tap] : 0.f;
  dst[idx] = (_Float16)v;
}

// fp32 (P,C) contiguous rows -> f16 (P,Cpad), zero-padded channels
__global__ void k_rows_to_f16(const float* __restrict__ s, _Float16* __restrict__ d,
                              int P, int C, int Cpad) {
  int i = blockIdx.x * 256 + threadIdx.x;
  if (i >= P * Cpad) return;
  int c = i % Cpad, p = i / Cpad;
  d[i] = (_Float16)((c < C) ? s[(size_t)p * C + c] : 0.f);
}

// fp32 channel-major (C,P) -> f16 pixel-major (P,Cpad)
__global__ void k_chw_to_f16(const float* __restrict__ s, _Float16* __restrict__ d,
                             int P, int C, int Cpad) {
  int i = blockIdx.x * 256 + threadIdx.x;
  if (i >= P * Cpad) return;
  int c = i % Cpad, p = i / Cpad;
  d[i] = (_Float16)((c < C) ? s[(size_t)c * P + p] : 0.f);
}

// ------------------------------------------------ implicit-GEMM WMMA conv
// out[oc][pix] = sum_{tap,ci} wA[oc][ci][tap] * in[pix+tap][ci]  (+bias, relu)
// A = 16 oc x 32 K (pre-packed), B = 32 K x 16 pixels (contiguous f16 rows).
// W is a power of two: wshift = log2(W) (no integer division in the kernel).
// Each wave: 4 M-tiles (64 oc) x 16 pixels; 4 WMMAs per B-operand load.
template <int TAPS, bool RELU>
__global__ __launch_bounds__(256)
void k_conv_wmma(const _Float16* __restrict__ inb,
                 const _Float16* __restrict__ wA,
                 const float* __restrict__ bias,
                 float* __restrict__ out,
                 int H, int wshift, int Cinp, int Cout) {
  const int P = H << wshift;
  const int lane = threadIdx.x & 31;
  const int tile = blockIdx.x * (blockDim.x >> 5) + (threadIdx.x >> 5);
  const int ntile = P >> 4;
  if (tile >= ntile) return;                      // wave-uniform
  const int mt0 = blockIdx.y << 2;
  const int MT4 = (int)gridDim.y << 2;
  const int KT = Cinp >> 5;
  const int n = lane & 15, g = lane >> 4;
  int ty = 0, px0 = 0;
  if (TAPS == 9) {
    ty  = tile >> (wshift - 4);
    px0 = (tile << 4) & ((1 << wshift) - 1);
  }

  v8f acc[4] = {};
#pragma unroll
  for (int tap = 0; tap < TAPS; ++tap) {
    bool ok = true;
    size_t pixin = (size_t)((tile << 4) + n);
    if (TAPS == 9) {
      const int dy = tap / 3 - 1;
      const int dx = tap % 3 - 1;
      const int yy = ty + dy;
      const int xx = px0 + n + dx;
      ok = ((unsigned)yy < (unsigned)H) & ((unsigned)xx < (unsigned)(1 << wshift));
      pixin = ((size_t)yy << wshift) + xx;
    }
    const _Float16* bp  = inb + pixin * Cinp + g * 16;
    const _Float16* ap0 = wA + ((size_t)tap * MT4 + mt0) * KT * 512 + lane * 16;
    for (int kt = 0; kt < KT; ++kt) {
      v16h b = {};
      if (ok) b = *(const v16h*)(bp + kt * 32);   // EXEC all-ones again below
      const _Float16* ap = ap0 + (size_t)kt * 512;
#pragma unroll
      for (int j = 0; j < 4; ++j) {               // 4 independent WMMA chains
        v16h a = *(const v16h*)(ap + (size_t)j * KT * 512);
        acc[j] = __builtin_amdgcn_wmma_f32_16x16x32_f16(false, a, false, b,
                                                        (short)0, acc[j],
                                                        false, false);
      }
    }
  }
  // C/D layout: VGPR r -> M = r + 8*g, lane[3:0] -> N (pixel)
  const int pix = (tile << 4) + n;
#pragma unroll
  for (int j = 0; j < 4; ++j) {
#pragma unroll
    for (int r = 0; r < 8; ++r) {
      int oc = (mt0 + j) * 16 + r + g * 8;
      if (oc < Cout) {
        float v = acc[j][r] + bias[oc];
        if (RELU) v = fmaxf(v, 0.f);
        out[(size_t)oc * P + pix] = v;
      }
    }
  }
}

// ------------------------------------------------------- elementwise kernels
// LayerNorm over the 64-channel axis at each pixel (torch LayerNorm on c).
__global__ void k_ln(const float* __restrict__ x, const float* __restrict__ gg,
                     const float* __restrict__ bb, float* __restrict__ o, int P) {
  int p = blockIdx.x * 256 + threadIdx.x;
  if (p >= P) return;
  float mu = 0.f;
  for (int c = 0; c < 64; ++c) mu += x[(size_t)c * P + p];
  mu *= (1.f / 64.f);
  float var = 0.f;
  for (int c = 0; c < 64; ++c) { float d = x[(size_t)c * P + p] - mu; var += d * d; }
  var *= (1.f / 64.f);
  float inv = rsqrtf(var + 1e-5f);
  for (int c = 0; c < 64; ++c)
    o[(size_t)c * P + p] = (x[(size_t)c * P + p] - mu) * inv * gg[c] + bb[c];
}

// transpose(+pixel_shuffle r=2) fused gather: t1 (64,256,256) -> dst (64,128,512)
// mode 0 = sag (b c h w -> b h c w), mode 1 = cor (b c h w -> b w c h)
__global__ void k_cv_expand(const float* __restrict__ t1, float* __restrict__ dst,
                            int mode) {
  int idx = blockIdx.x * 256 + threadIdx.x;
  if (idx >= 64 * 128 * 512) return;
  int x = idx & 511, y = (idx >> 9) & 127, c2 = idx >> 16;
  int a = c2 * 4 + ((y & 1) << 1) + (x & 1);
  size_t src = (mode == 0)
             ? ((size_t)(y >> 1) * 65536 + (size_t)a * 256 + (x >> 1))
             : ((size_t)(y >> 1) * 65536 + (size_t)(x >> 1) * 256 + a);
  dst[idx] = t1[src];
}

// pixel_unshuffle + inverse transpose for both branches, summed, + skip
__global__ void k_cv_merge(const float* __restrict__ t5s, const float* __restrict__ t5c,
                           const float* __restrict__ skip, float* __restrict__ dst) {
  int idx = blockIdx.x * 256 + threadIdx.x;
  if (idx >= 64 * 256 * 256) return;
  int c = idx >> 16, h = (idx >> 8) & 255, w = idx & 255;
  float s = t5s[(size_t)(h >> 2) * 65536 + (size_t)(c * 2 + ((h >> 1) & 1)) * 512 +
                (w * 2 + (h & 1))];
  float r = t5c[(size_t)(w >> 2) * 65536 + (size_t)(c * 2 + ((w >> 1) & 1)) * 512 +
                (h * 2 + (w & 1))];
  dst[idx] = s + r + skip[idx];
}

// causal depthwise conv (d_conv=4, left pad 3) + bias + SiLU on xz[0:512]
__global__ void k_dwconv_silu(const float* __restrict__ xz, const float* __restrict__ w,
                              const float* __restrict__ cb, float* __restrict__ o,
                              int PM) {
  int idx = blockIdx.x * 256 + threadIdx.x;
  if (idx >= 512 * PM) return;
  int p = idx % PM, e = idx / PM;
  int l = p & 255;
  float acc = cb[e];
#pragma unroll
  for (int k = 0; k < 4; ++k) {
    int ls = l - 3 + k;
    if (ls >= 0) acc += w[e * 4 + k] * xz[(size_t)idx - 3 + k];
  }
  o[idx] = silu_f(acc);
}

__global__ void k_softplus(float* __restrict__ d, int n) {
  int i = blockIdx.x * 256 + threadIdx.x;
  if (i >= n) return;
  float v = d[i];
  d[i] = (v > 20.f) ? v : log1pf(__expf(v));
}

__global__ void k_negexp(const float* __restrict__ s, float* __restrict__ d, int n) {
  int i = blockIdx.x * 256 + threadIdx.x;
  if (i < n) d[i] = -__expf(s[i]);
}

// selective scan: thread = (b,e); 16 states in registers; Bt/Ct via LDS bcast
__global__ void k_scan(float* __restrict__ dy /* delta in, y out, (512,PM) */,
                       const float* __restrict__ xs, const float* __restrict__ xdbl,
                       const float* __restrict__ A, int PM) {
  __shared__ float sB[16], sC[16];
  int b = blockIdx.x >> 1;
  int e = ((blockIdx.x & 1) << 8) + threadIdx.x;
  float Ae[16], h[16];
#pragma unroll
  for (int s = 0; s < 16; ++s) { Ae[s] = A[e * 16 + s]; h[s] = 0.f; }
  const size_t base = (size_t)b * 256;
  for (int l = 0; l < 256; ++l) {
    size_t p = base + l;
    if (threadIdx.x < 32) {
      int t = threadIdx.x;
      if (t < 16) sB[t] = xdbl[(size_t)(16 + t) * PM + p];
      else        sC[t - 16] = xdbl[(size_t)(32 + t - 16) * PM + p];
    }
    __syncthreads();
    float d  = dy[(size_t)e * PM + p];
    float xv = xs[(size_t)e * PM + p];
    __builtin_prefetch(&dy[(size_t)e * PM + p + 8], 0, 0);  // global_prefetch_b8
    float dx = d * xv, y = 0.f;
#pragma unroll
    for (int s = 0; s < 16; ++s) {
      h[s] = __expf(d * Ae[s]) * h[s] + dx * sB[s];
      y += h[s] * sC[s];
    }
    dy[(size_t)e * PM + p] = y;   // in-place: this element already consumed
    __syncthreads();
  }
}

// y = (y + xs*D[e]) * silu(z), in place
__global__ void k_ypost(float* __restrict__ y, const float* __restrict__ xs,
                        const float* __restrict__ D, const float* __restrict__ z,
                        int n, int PM) {
  int i = blockIdx.x * 256 + threadIdx.x;
  if (i >= n) return;
  int e = i / PM;
  y[i] = (y[i] + xs[i] * D[e]) * silu_f(z[i]);
}

// res2[c,h,w] = a0[c,h,w] + mout[w][c*256+h]   (mout is (256, 16384) ch-major)
__global__ void k_add_mamba(const float* __restrict__ a0, const float* __restrict__ mout,
                            float* __restrict__ res2, int n) {
  int i = blockIdx.x * 256 + threadIdx.x;
  if (i >= n) return;
  res2[i] = a0[i] + mout[(size_t)(i & 255) * 16384 + (i >> 8)];
}

__global__ void k_add2(float* __restrict__ a, const float* __restrict__ b, int n) {
  int i = blockIdx.x * 256 + threadIdx.x;
  if (i < n) a[i] += b[i];
}

// concat([a0,a1], channel) -> f16 pixel-major rows of 128
__global__ void k_concat_f16(const float* __restrict__ a0, const float* __restrict__ a1,
                             _Float16* __restrict__ d, int P) {
  int i = blockIdx.x * 256 + threadIdx.x;
  if (i >= P * 128) return;
  int c = i & 127, p = i >> 7;
  d[i] = (_Float16)((c < 64) ? a0[(size_t)c * P + p] : a1[(size_t)(c - 64) * P + p]);
}

// NHWC output; even channels overwritten by the input (out[:, ::2] = xp)
__global__ void k_final(const float* __restrict__ tout, const float* __restrict__ x,
                        float* __restrict__ out) {
  int i = blockIdx.x * 256 + threadIdx.x;
  if (i >= 65536 * 16) return;
  int c = i & 15, p = i >> 4;
  out[i] = ((c & 1) == 0) ? x[(size_t)p * 8 + (c >> 1)]
                          : tout[(size_t)c * 65536 + p];
}

// ---------------------------------------------------------------- launcher
extern "C" void kernel_launch(void* const* d_in, const int* in_sizes, int n_in,
                              void* d_out, int out_size, void* d_ws, size_t ws_size,
                              hipStream_t stream) {
  (void)in_sizes; (void)n_in; (void)out_size; (void)ws_size;
  const int P0 = 65536, PM = 16384;

  const float* x    = (const float*)d_in[0];
  const float* hw1  = (const float*)d_in[1];
  const float* hb1  = (const float*)d_in[2];
  const float* hw2  = (const float*)d_in[3];
  const float* hb2  = (const float*)d_in[4];
  const int A0 = 5, A1 = 19;
  const float* min_w   = (const float*)d_in[33];
  const float* mconv_w = (const float*)d_in[34];
  const float* mconv_b = (const float*)d_in[35];
  const float* mxproj  = (const float*)d_in[36];
  const float* mdt_w   = (const float*)d_in[37];
  const float* mdt_b   = (const float*)d_in[38];
  const float* mA_log  = (const float*)d_in[39];
  const float* mD      = (const float*)d_in[40];
  const float* mout_w  = (const float*)d_in[41];
  const float* fw  = (const float*)d_in[42];
  const float* fb  = (const float*)d_in[43];
  const float* tw1 = (const float*)d_in[44];
  const float* tb1 = (const float*)d_in[45];
  const float* tw2 = (const float*)d_in[46];
  const float* tb2 = (const float*)d_in[47];

  // ---- workspace bump allocator (all buffers fit in L2-resident scratch)
  char* wp = (char*)d_ws;
  auto alloc = [&](size_t bytes) -> void* {
    void* r = (void*)wp;
    wp += (bytes + 255) & ~(size_t)255;
    return r;
  };
  float* zeros = (float*)alloc(1024 * 4);
  _Float16* ph1 = (_Float16*)alloc((size_t)9 * 4 * 1 * 512 * 2);
  _Float16* ph2 = (_Float16*)alloc((size_t)9 * 4 * 2 * 512 * 2);
  _Float16* cvp[2][6];
  for (int a = 0; a < 2; ++a)
    for (int br = 0; br < 2; ++br) {
      cvp[a][br * 3 + 0] = (_Float16*)alloc((size_t)1 * 4 * 2 * 512 * 2);
      cvp[a][br * 3 + 1] = (_Float16*)alloc((size_t)9 * 4 * 2 * 512 * 2);
      cvp[a][br * 3 + 2] = (_Float16*)alloc((size_t)9 * 4 * 2 * 512 * 2);
    }
  _Float16* pin   = (_Float16*)alloc((size_t)64 * 8 * 512 * 2);
  _Float16* pxp   = (_Float16*)alloc((size_t)4 * 16 * 512 * 2);   // MT 3 -> 4
  _Float16* pdt   = (_Float16*)alloc((size_t)32 * 1 * 512 * 2);
  _Float16* pout  = (_Float16*)alloc((size_t)16 * 16 * 512 * 2);
  _Float16* pfuse = (_Float16*)alloc((size_t)4 * 4 * 512 * 2);
  _Float16* pt1   = (_Float16*)alloc((size_t)9 * 4 * 2 * 512 * 2);
  _Float16* pt2   = (_Float16*)alloc((size_t)9 * 4 * 2 * 512 * 2); // MT 1 -> 4
  float* Abuf = (float*)alloc(8192 * 4);
  _Float16* fb16 = (_Float16*)alloc((size_t)PM * 512 * 2);   // 16.8 MB, reused
  float* x_head = (float*)alloc((size_t)64 * P0 * 4);
  float* a0     = (float*)alloc((size_t)64 * P0 * 4);
  float* res2   = (float*)alloc((size_t)64 * P0 * 4);
  float* a1     = (float*)alloc((size_t)64 * P0 * 4);
  char* blob = (char*)alloc((size_t)132 << 20);              // cv/mamba union
  float* t_xn  = (float*)(blob);
  float* t_a   = (float*)(blob + ((size_t)16 << 20));
  float* t_b   = (float*)(blob + ((size_t)32 << 20));
  float* t_sag = (float*)(blob + ((size_t)48 << 20));
  float* t_cor = (float*)(blob + ((size_t)64 << 20));
  float* xz      = (float*)(blob);                           // (1024, PM)
  float* xs_post = (float*)(blob + ((size_t)64 << 20));      // (512, PM)
  float* xdbl    = (float*)(blob + ((size_t)96 << 20));      // (48, PM)
  float* delta   = (float*)(blob + ((size_t)100 << 20));     // (512, PM), -> y

  auto g1 = [](int n) { return dim3((unsigned)((n + 255) / 256)); };
  auto mt4_of = [](int Cout) { int MT = (Cout + 15) >> 4; return (MT + 3) & ~3; };
  auto pack = [&](const float* w, _Float16* dst, int Cout, int Cin, int taps, int Cinp) {
    int MT4 = mt4_of(Cout), KT = Cinp / 32;
    int total = taps * MT4 * KT * 512;
    k_pack_a<<<g1(total), dim3(256), 0, stream>>>(w, dst, Cout, Cin, taps, MT4, KT);
  };
  auto conv = [&](int taps, bool relu, const _Float16* inb, const _Float16* wA,
                  const float* bias, float* out, int H, int wshift, int Cinp, int Cout) {
    int ntile = (H << wshift) >> 4;
    dim3 grd((unsigned)((ntile + 7) >> 3), (unsigned)(mt4_of(Cout) >> 2));
    dim3 blk(256);
    if (taps == 9) {
      if (relu) k_conv_wmma<9, true><<<grd, blk, 0, stream>>>(inb, wA, bias, out, H, wshift, Cinp, Cout);
      else      k_conv_wmma<9, false><<<grd, blk, 0, stream>>>(inb, wA, bias, out, H, wshift, Cinp, Cout);
    } else {
      k_conv_wmma<1, false><<<grd, blk, 0, stream>>>(inb, wA, bias, out, H, wshift, Cinp, Cout);
    }
  };

  // ---- constant prep
  k_zero<<<g1(1024), dim3(256), 0, stream>>>(zeros, 1024);
  pack(hw1, ph1, 64, 8, 9, 32);
  pack(hw2, ph2, 64, 64, 9, 64);
  for (int a = 0; a < 2; ++a) {
    int base = a ? A1 : A0;
    for (int br = 0; br < 2; ++br) {
      int i0 = base + 2 + br * 6;
      pack((const float*)d_in[i0 + 0], cvp[a][br * 3 + 0], 64, 64, 1, 64);
      pack((const float*)d_in[i0 + 2], cvp[a][br * 3 + 1], 64, 64, 9, 64);
      pack((const float*)d_in[i0 + 4], cvp[a][br * 3 + 2], 64, 64, 9, 64);
    }
  }
  pack(min_w, pin, 1024, 256, 1, 256);
  pack(mxproj, pxp, 48, 512, 1, 512);
  pack(mdt_w, pdt, 512, 16, 1, 32);
  pack(mout_w, pout, 256, 512, 1, 512);
  pack(fw, pfuse, 64, 128, 1, 128);
  pack(tw1, pt1, 64, 64, 9, 64);
  pack(tw2, pt2, 16, 64, 9, 64);
  k_negexp<<<g1(8192), dim3(256), 0, stream>>>(mA_log, Abuf, 8192);

  // ---- head
  k_rows_to_f16<<<g1(P0 * 32), dim3(256), 0, stream>>>(x, fb16, P0, 8, 32);
  conv(9, true, fb16, ph1, hb1, t_a, 256, 8, 32, 64);
  k_chw_to_f16<<<g1(P0 * 64), dim3(256), 0, stream>>>(t_a, fb16, P0, 64, 64);
  conv(9, false, fb16, ph2, hb2, x_head, 256, 8, 64, 64);

  // ---- cross-view block
  auto cross_view = [&](const float* src, const float* skip, float* dst, int base,
                        _Float16* const* pk) {
    k_ln<<<g1(P0), dim3(256), 0, stream>>>(src, (const float*)d_in[base],
                                           (const float*)d_in[base + 1], t_xn, P0);
    for (int br = 0; br < 2; ++br) {
      int i0 = base + 2 + br * 6;
      const float* b1 = (const float*)d_in[i0 + 1];
      const float* b2 = (const float*)d_in[i0 + 3];
      const float* b3 = (const float*)d_in[i0 + 5];
      float* tout = br ? t_cor : t_sag;
      k_chw_to_f16<<<g1(P0 * 64), dim3(256), 0, stream>>>(t_xn, fb16, P0, 64, 64);
      conv(1, false, fb16, pk[br * 3 + 0], b1, t_a, 256, 8, 64, 64);
      k_cv_expand<<<g1(P0 * 64), dim3(256), 0, stream>>>(t_a, t_b, br);
      k_chw_to_f16<<<g1(P0 * 64), dim3(256), 0, stream>>>(t_b, fb16, P0, 64, 64);
      conv(9, true, fb16, pk[br * 3 + 1], b2, t_a, 128, 9, 64, 64);
      k_chw_to_f16<<<g1(P0 * 64), dim3(256), 0, stream>>>(t_a, fb16, P0, 64, 64);
      conv(9, false, fb16, pk[br * 3 + 2], b3, tout, 128, 9, 64, 64);
    }
    k_cv_merge<<<g1(P0 * 64), dim3(256), 0, stream>>>(t_sag, t_cor, skip, dst);
  };

  cross_view(x_head, x_head, a0, A0, cvp[0]);   // a0 = cv(x_head) + x_head

  // ---- Mamba (I2Block): u = a0 viewed as (16384, 256) rows
  k_rows_to_f16<<<g1(PM * 256), dim3(256), 0, stream>>>(a0, fb16, PM, 256, 256);
  conv(1, false, fb16, pin, zeros, xz, 1, 14, 256, 1024);                 // in_proj
  k_dwconv_silu<<<g1(512 * PM), dim3(256), 0, stream>>>(xz, mconv_w, mconv_b, xs_post, PM);
  k_chw_to_f16<<<g1(PM * 512), dim3(256), 0, stream>>>(xs_post, fb16, PM, 512, 512);
  conv(1, false, fb16, pxp, zeros, xdbl, 1, 14, 512, 48);                 // x_proj
  k_chw_to_f16<<<g1(PM * 32), dim3(256), 0, stream>>>(xdbl, fb16, PM, 16, 32);
  conv(1, false, fb16, pdt, mdt_b, delta, 1, 14, 32, 512);                // dt_proj
  k_softplus<<<g1(512 * PM), dim3(256), 0, stream>>>(delta, 512 * PM);
  k_scan<<<dim3(128), dim3(256), 0, stream>>>(delta, xs_post, xdbl, Abuf, PM);
  k_ypost<<<g1(512 * PM), dim3(256), 0, stream>>>(delta, xs_post, mD,
                                                  xz + (size_t)512 * PM, 512 * PM, PM);
  k_chw_to_f16<<<g1(PM * 512), dim3(256), 0, stream>>>(delta, fb16, PM, 512, 512);
  conv(1, false, fb16, pout, zeros, xz /* mout */, 1, 14, 512, 256);      // out_proj
  k_add_mamba<<<g1(64 * P0), dim3(256), 0, stream>>>(a0, xz, res2, 64 * P0);

  // ---- second cross-view
  cross_view(res2, res2, a1, A1, cvp[1]);       // a1 = cv(res2) + res2

  // ---- fuse + residual
  k_concat_f16<<<g1(P0 * 128), dim3(256), 0, stream>>>(a0, a1, fb16, P0);
  conv(1, false, fb16, pfuse, fb, t_a, 1, 16, 128, 64);
  k_add2<<<g1(64 * P0), dim3(256), 0, stream>>>(t_a, x_head, 64 * P0);

  // ---- tail
  k_chw_to_f16<<<g1(P0 * 64), dim3(256), 0, stream>>>(t_a, fb16, P0, 64, 64);
  conv(9, true, fb16, pt1, tb1, t_b, 256, 8, 64, 64);
  k_chw_to_f16<<<g1(P0 * 64), dim3(256), 0, stream>>>(t_b, fb16, P0, 64, 64);
  conv(9, false, fb16, pt2, tb2, t_xn, 256, 8, 64, 16);

  // ---- NHWC output with even-channel passthrough of the input
  k_final<<<g1(P0 * 16), dim3(256), 0, stream>>>(t_xn, x, (float*)d_out);
}
